// FeatureAdaption_42210938585592
// MI455X (gfx1250) — compile-verified, matrix operand order checked by reading, not run on hardware
//
#include <hip/hip_runtime.h>
#include <math.h>

// ---------------------------------------------------------------------------
// Deformable-conv "FeatureAdaption" for MI455X (gfx1250, wave32, WMMA).
//   B=8, C=256, H=W=64, G=4 (Cg=64), Co=256, K=3x3.
// Implicit GEMM: M=Co=256, N=pixels=32768, Kdim=C*9=2304, bf16 A/B, f32 acc,
// via v_wmma_f32_16x16x32_bf16. Weight panels staged with CDNA5 async
// global->LDS DMA (ASYNCcnt) overlapped with the bilinear gather.
// ---------------------------------------------------------------------------

typedef __attribute__((ext_vector_type(16))) __bf16 v16bf;
typedef __attribute__((ext_vector_type(8)))  __bf16 v8bf;
typedef __attribute__((ext_vector_type(8)))  float  v8f;

#define BB   8
#define CIN  256
#define CO   256
#define GG   4
#define CG   64
#define HW   4096          // 64*64
#define NT   32            // pixels per block (N tile)
#define LDA  72            // padded LDS row stride in bf16 elems (144 B = 9*16B)

// --- prep 1: transpose x [B,C,H,W] -> xt [B,H,W,C] (f32) ----------------
__global__ __launch_bounds__(256) void xpose_kernel(const float* __restrict__ x,
                                                    float* __restrict__ xt, int n) {
  int i = blockIdx.x * 256 + threadIdx.x;
  if (i >= n) return;
  int c = i & 255;
  int p = (i >> 8) & (HW - 1);
  int b = i >> 20;                       // 4096*256 = 2^20 per image
  xt[i] = x[(((size_t)b * CIN + c) << 12) + p];
}

// --- prep 2: w_d [Co,C,3,3] f32 -> wt bf16 [G][9][Co][Cg] ---------------
__global__ __launch_bounds__(256) void wprep_kernel(const float* __restrict__ wd,
                                                    __bf16* __restrict__ wt, int n) {
  int i = blockIdx.x * 256 + threadIdx.x;
  if (i >= n) return;
  int c = i & 63;
  int o = (i >> 6) & 255;
  int t = (i >> 14) % 9;
  int g = i / (64 * 256 * 9);
  wt[i] = (__bf16)wd[(size_t)o * 2304 + (size_t)(g * 64 + c) * 9 + t];
}

#define LOAD8(dst, ptr)                                                   \
  { float4 _a = ((const float4*)(ptr))[0];                                \
    float4 _b = ((const float4*)(ptr))[1];                                \
    dst[0]=_a.x; dst[1]=_a.y; dst[2]=_a.z; dst[3]=_a.w;                   \
    dst[4]=_b.x; dst[5]=_b.y; dst[6]=_b.z; dst[7]=_b.w; }

// --- main: offsets + bilinear sampling + WMMA GEMM ----------------------
__global__ __launch_bounds__(256) void fadapt_kernel(
    const float* __restrict__ xt,      // [B,H,W,C]
    const __bf16* __restrict__ wt,     // [G,9,Co,Cg] bf16
    const float* __restrict__ whp,     // [B,2G,H,W]
    const float* __restrict__ woff,    // [G,18,2]
    const float* __restrict__ boff,    // [G*18]
    const float* __restrict__ bd,      // [Co]
    float* __restrict__ out)           // [B,Co,H,W]
{
  __shared__ __attribute__((aligned(16))) __bf16 Ald[CO * LDA];  // weight panel
  __shared__ __attribute__((aligned(16))) __bf16 Bld[NT * LDA];  // sample panel

  const int tid = threadIdx.x;
  const int b   = blockIdx.x >> 7;            // 128 blocks per image
  const int p0  = (blockIdx.x & 127) * NT;

  // sampling role: (pixel, channel-octet)
  const int pix = tid >> 3;                   // 0..31
  const int oct = tid & 7;                    // 0..7 -> channels oct*8..+8
  const int p   = p0 + pix;
  const int ph  = p >> 6;
  const int pw  = p & 63;

  // wmma role: 8 waves = 4 (M) x 2 (N)
  const int lane  = tid & 31;
  const int wid   = tid >> 5;
  const int wm    = (wid & 3) * 64;           // M base (4 tiles of 16)
  const int wn    = (wid >> 2) * 16;          // N base
  const int lr    = lane & 15;
  const int halfk = (lane >> 4) * 8;          // K-run select per 16-bit layout

  // LDS byte offset of this thread's weight row (flat shared addr low bits
  // == wave-relative LDS offset per the ISA flat->LDS aperture mapping).
  const unsigned ldsA = (unsigned)(uintptr_t)(&Ald[tid * LDA]);

  v8f acc[4] = {};

  for (int g = 0; g < GG; ++g) {
    const float wh0 = whp[(((size_t)b * 8 + 2 * g) << 12) + p];
    const float wh1 = whp[(((size_t)b * 8 + 2 * g + 1) << 12) + p];
    const float* xbase = xt + ((size_t)b << 20) + g * CG + oct * 8;

    for (int t = 0; t < 9; ++t) {
      // ---- stage A panel via async global->LDS DMA (ASYNCcnt) ----------
      // Thread tid owns output-channel row o = tid: 64 bf16 = 128 B = 8 x b128.
      {
        unsigned long long ga =
            (unsigned long long)(uintptr_t)(wt + (((size_t)(g * 9 + t) * 256 + tid) << 6));
#pragma unroll
        for (int j = 0; j < 8; ++j) {
          asm volatile("global_load_async_to_lds_b128 %0, %1, off"
                       :
                       : "v"(ldsA + j * 16), "v"(ga + (unsigned long long)(j * 16))
                       : "memory");
        }
      }

      // ---- offsets + bilinear sample -> B panel (overlaps the DMA) -----
      {
        const int   o2  = g * 18 + 2 * t;
        const float a00 = woff[o2 * 2 + 0], a01 = woff[o2 * 2 + 1];
        const float a10 = woff[o2 * 2 + 2], a11 = woff[o2 * 2 + 3];
        const float oy  = wh0 * a00 + wh1 * a01 + boff[o2];
        const float ox  = wh0 * a10 + wh1 * a11 + boff[o2 + 1];
        const float py  = oy + (float)(t / 3 - 1 + ph);
        const float px  = ox + (float)(t % 3 - 1 + pw);
        const float fy  = floorf(py), fx = floorf(px);
        const float wy  = py - fy,    wx = px - fx;
        const int y0 = (int)fy, x0 = (int)fx, y1 = y0 + 1, x1 = x0 + 1;
        const int yi0 = (y0 >= 0) & (y0 < 64), yi1 = (y1 >= 0) & (y1 < 64);
        const int xi0 = (x0 >= 0) & (x0 < 64), xi1 = (x1 >= 0) & (x1 < 64);
        float w00 = (1.f - wy) * (1.f - wx) * (float)(yi0 & xi0);
        float w01 = (1.f - wy) * wx         * (float)(yi0 & xi1);
        float w10 = wy * (1.f - wx)         * (float)(yi1 & xi0);
        float w11 = wy * wx                 * (float)(yi1 & xi1);
        const int y0c = min(max(y0, 0), 63), y1c = min(max(y1, 0), 63);
        const int x0c = min(max(x0, 0), 63), x1c = min(max(x1, 0), 63);
        const float* q00 = xbase + (size_t)((y0c << 6) + x0c) * CIN;
        const float* q01 = xbase + (size_t)((y0c << 6) + x1c) * CIN;
        const float* q10 = xbase + (size_t)((y1c << 6) + x0c) * CIN;
        const float* q11 = xbase + (size_t)((y1c << 6) + x1c) * CIN;
        float r00[8], r01[8], r10[8], r11[8];
        LOAD8(r00, q00); LOAD8(r01, q01); LOAD8(r10, q10); LOAD8(r11, q11);
        v8bf sv;
#pragma unroll
        for (int j = 0; j < 8; ++j)
          sv[j] = (__bf16)(w00 * r00[j] + w01 * r01[j] + w10 * r10[j] + w11 * r11[j]);
        *(v8bf*)&Bld[pix * LDA + oct * 8] = sv;
      }

      // All async weight DMAs into Ald must land before anyone reads it.
      asm volatile("s_wait_asynccnt 0x0" ::: "memory");
      __syncthreads();

      // ---- WMMA over this 64-deep K panel (2 steps of 32) ----
#pragma unroll
      for (int ks = 0; ks < 64; ks += 32) {
        const int bk = ks + halfk;
        const __bf16* bp = &Bld[(wn + lr) * LDA + bk];
        v8bf blo = *(const v8bf*)bp;
        v8bf bhi = *(const v8bf*)(bp + 16);
        v16bf bf;
#pragma unroll
        for (int j = 0; j < 8; ++j) { bf[j] = blo[j]; bf[j + 8] = bhi[j]; }
#pragma unroll
        for (int mt = 0; mt < 4; ++mt) {
          const __bf16* ap = &Ald[(wm + mt * 16 + lr) * LDA + bk];
          v8bf alo = *(const v8bf*)ap;
          v8bf ahi = *(const v8bf*)(ap + 16);
          v16bf af;
#pragma unroll
          for (int j = 0; j < 8; ++j) { af[j] = alo[j]; af[j + 8] = ahi[j]; }
          acc[mt] = __builtin_amdgcn_wmma_f32_16x16x32_bf16(
              false, af, false, bf, (short)0, acc[mt], false, false);
        }
      }
      __syncthreads();
    }
  }

  // ---- epilogue: C/D layout (VGPR r: M = halfk + r), add bias ----
#pragma unroll
  for (int mt = 0; mt < 4; ++mt) {
#pragma unroll
    for (int r = 0; r < 8; ++r) {
      const int m = wm + mt * 16 + halfk + r;
      out[(((size_t)b * CO + m) << 12) + p0 + wn + lr] = acc[mt][r] + bd[m];
    }
  }
}

extern "C" void kernel_launch(void* const* d_in, const int* in_sizes, int n_in,
                              void* d_out, int out_size, void* d_ws, size_t ws_size,
                              hipStream_t stream) {
  const float* x    = (const float*)d_in[0];
  const float* whp  = (const float*)d_in[1];
  const float* woff = (const float*)d_in[2];
  const float* boff = (const float*)d_in[3];
  const float* wd   = (const float*)d_in[4];
  const float* bd   = (const float*)d_in[5];
  float* out = (float*)d_out;

  float*  xt = (float*)d_ws;                                   // 33.55 MB
  __bf16* wt = (__bf16*)((char*)d_ws + (size_t)BB * HW * CIN * 4);  // +1.18 MB

  const int nx = BB * CIN * HW;                // 8,388,608
  xpose_kernel<<<nx / 256, 256, 0, stream>>>(x, xt, nx);
  const int nw = GG * 9 * CO * CG;             // 589,824
  wprep_kernel<<<nw / 256, 256, 0, stream>>>(wd, wt, nw);
  fadapt_kernel<<<BB * HW / NT, 256, 0, stream>>>(xt, wt, whp, woff, boff, bd, out);
}
